// MoELayer_32684701123237
// MI455X (gfx1250) — compile-verified, matrix operand order checked by reading, not run on hardware
//
#include <hip/hip_runtime.h>

// ---------------------------------------------------------------------------
// MoE layer for MI455X (gfx1250, wave32, WMMA bf16 16x16x32 + TDM async tiles)
// ---------------------------------------------------------------------------

typedef __bf16 bf16_t;
typedef bf16_t v16bf __attribute__((ext_vector_type(16)));
typedef bf16_t v8bf  __attribute__((ext_vector_type(8)));
typedef float  v8f   __attribute__((ext_vector_type(8)));

typedef unsigned int u32x4 __attribute__((ext_vector_type(4)));
typedef int          i32x4 __attribute__((ext_vector_type(4)));
typedef int          i32x8 __attribute__((ext_vector_type(8)));

#define T_TOK   4096
#define DMODEL  1024
#define DFF     2048
#define NEXP    8
#define FSHARED 4096

#define BM 64
#define BN 128
#define BK 32
#define LDT 40   // LDS row stride in bf16 elems (64B data + 16B pad, 16B aligned)

// ---------------------------------------------------------------------------
// helpers
// ---------------------------------------------------------------------------
__device__ inline v8f wmma_bf16(v16bf a, v16bf b, v8f c) {
    // D = A(16x32 bf16) * B(32x16 bf16) + C(16x16 f32)
    return __builtin_amdgcn_wmma_f32_16x16x32_bf16(
        false, a, false, b, (short)0, c, false, false);
}

// Fragment load from an LDS tile stored row-major (rows of 32 bf16, stride LDT).
// 16-bit A/B layout: lane r=lane&15 is the row (A) / col (B); kh=lane>>4 picks
// K halves {kh*8..kh*8+7, 16+kh*8..16+kh*8+7}. Two contiguous 16B reads.
__device__ inline v16bf frag16(const bf16_t* rowp, int kh) {
    v8bf lo = *(const v8bf*)(rowp + kh * 8);
    v8bf hi = *(const v8bf*)(rowp + 16 + kh * 8);
    return __builtin_shufflevector(lo, hi, 0,1,2,3,4,5,6,7,8,9,10,11,12,13,14,15);
}

__device__ inline v8bf zero_v8bf() {
    v8bf z;
#pragma unroll
    for (int i = 0; i < 8; ++i) z[i] = (bf16_t)0.0f;
    return z;
}
__device__ inline v8f zero_v8f() {
    v8f z;
#pragma unroll
    for (int i = 0; i < 8; ++i) z[i] = 0.0f;
    return z;
}

// Low 32 bits of a generic LDS pointer == byte offset within workgroup LDS.
__device__ inline unsigned lds_off(const void* p) {
    return (unsigned)(unsigned long long)p;
}

// ---------------------------------------------------------------------------
// Tensor Data Mover: DMA a [tileRows x BK] bf16 tile (row stride K elems) from
// global into LDS, inserting 16B of padding after each 64B row so the LDS
// layout matches the LDT=40 stride used by frag16.
// D# layout per CDNA5 ISA ch.8 (group0 128b, group1 256b; groups 2/3 unused).
// ---------------------------------------------------------------------------
__device__ inline void tdm_load_tile(const bf16_t* gtile, unsigned ldsByteOff,
                                     int K, int tileRows) {
    unsigned long long ga = (unsigned long long)gtile;
    u32x4 g0;
    g0[0] = 1u;                                   // count=1, user mode
    g0[1] = ldsByteOff;                           // lds_addr [63:32]
    g0[2] = (unsigned)(ga & 0xFFFFFFFFu);         // global_addr lo [95:64]
    g0[3] = (unsigned)((ga >> 32) & 0x1FFFFFFu)   // global_addr hi [120:96]
          | (2u << 30);                           // type=2 ("image") [127:126]

    unsigned td0 = (unsigned)K;                   // tensor_dim0 (elements)
    unsigned td1 = (unsigned)tileRows;            // tensor_dim1 (rows from tile start)
    i32x8 g1;
    g1[0] = (int)((1u << 16)     // data_size = 2 bytes
                | (1u << 20)     // pad_enable
                | (3u << 22)     // pad_interval: 16 DWORDs (64B) between pads
                | (3u << 25));   // pad_amount: 4 DWORDs (16B)
    g1[1] = (int)((td0 & 0xFFFFu) << 16);                              // [63:48]=td0.lo
    g1[2] = (int)(((td0 >> 16) & 0xFFFFu) | ((td1 & 0xFFFFu) << 16));  // td0.hi | td1.lo
    g1[3] = (int)(((td1 >> 16) & 0xFFFFu) | ((unsigned)BK << 16));     // td1.hi | tile_dim0
    g1[4] = (int)((unsigned)tileRows & 0xFFFFu);                       // tile_dim1 (tile_dim2=0)
    g1[5] = (int)((unsigned)K);                                        // tensor_dim0_stride lo32
    g1[6] = 0;                                                         // stride hi | dim1_stride lo
    g1[7] = 0;
    i32x4 gz = {0, 0, 0, 0};
#if defined(__clang_major__) && __clang_major__ >= 23
    i32x8 gz8 = {0, 0, 0, 0, 0, 0, 0, 0};
    __builtin_amdgcn_tensor_load_to_lds(g0, g1, gz, gz, gz8, 0);
#else
    __builtin_amdgcn_tensor_load_to_lds(g0, g1, gz, gz, 0);
#endif
}

// ---------------------------------------------------------------------------
// Gating: comb[t][e] = renormalized top-2 of softmax(x @ Wg), zeros elsewhere
// ---------------------------------------------------------------------------
__global__ __launch_bounds__(256) void gate_kernel(
    const float* __restrict__ x, const float* __restrict__ Wg,
    float* __restrict__ comb) {
    int lane = threadIdx.x & 31;
    int tok  = blockIdx.x * 8 + (threadIdx.x >> 5);
    const float* xr = x + (size_t)tok * DMODEL;

    float acc[NEXP];
#pragma unroll
    for (int e = 0; e < NEXP; ++e) acc[e] = 0.0f;

    for (int d = lane; d < DMODEL; d += 32) {
        float xv = xr[d];
        const float* wr = Wg + (size_t)d * NEXP;
#pragma unroll
        for (int e = 0; e < NEXP; ++e) acc[e] += xv * wr[e];
    }
#pragma unroll
    for (int e = 0; e < NEXP; ++e) {
#pragma unroll
        for (int off = 16; off > 0; off >>= 1)
            acc[e] += __shfl_down(acc[e], off, 32);
    }
    if (lane == 0) {
        float m = acc[0];
#pragma unroll
        for (int e = 1; e < NEXP; ++e) m = fmaxf(m, acc[e]);
        float es[NEXP], sum = 0.0f;
#pragma unroll
        for (int e = 0; e < NEXP; ++e) { es[e] = __expf(acc[e] - m); sum += es[e]; }
        float sc[NEXP];
#pragma unroll
        for (int e = 0; e < NEXP; ++e) sc[e] = es[e] / sum;
        int i1 = 0;
#pragma unroll
        for (int e = 1; e < NEXP; ++e) if (sc[e] > sc[i1]) i1 = e;
        int i2 = (i1 == 0) ? 1 : 0;
#pragma unroll
        for (int e = 0; e < NEXP; ++e)
            if (e != i1 && sc[e] > sc[i2]) i2 = e;
        float a = sc[i1], b = sc[i2];
        float mm = fmaxf(a, b);
        float e1 = __expf(a - mm), e2 = __expf(b - mm);
        float w1 = e1 / (e1 + e2), w2 = e2 / (e1 + e2);
        float row[NEXP];
#pragma unroll
        for (int e = 0; e < NEXP; ++e) row[e] = 0.0f;
        row[i1] = w1; row[i2] = w2;
        float* cr = comb + (size_t)tok * NEXP;
#pragma unroll
        for (int e = 0; e < NEXP; ++e) cr[e] = row[e];
    }
}

// ---------------------------------------------------------------------------
// Deterministic per-expert token lists (one wave per expert, shfl scan).
// ---------------------------------------------------------------------------
__global__ __launch_bounds__(32) void build_lists(
    const float* __restrict__ comb, int* __restrict__ idx,
    float* __restrict__ wl, int* __restrict__ sl, int* __restrict__ counts) {
    int e = blockIdx.x;
    int lane = threadIdx.x;
    int count = 0;
    for (int base = 0; base < T_TOK; base += 32) {
        int t = base + lane;
        float w = comb[(size_t)t * NEXP + e];
        int f = (w > 0.0f) ? 1 : 0;
        int scan = f;
#pragma unroll
        for (int off = 1; off < 32; off <<= 1) {
            int n = __shfl_up(scan, off, 32);
            if (lane >= off) scan += n;
        }
        int total = __shfl(scan, 31, 32);
        if (f) {
            int pos = count + scan - 1;
            idx[(size_t)e * T_TOK + pos] = t;
            wl [(size_t)e * T_TOK + pos] = w;
            int s = 0;
            for (int e2 = 0; e2 < e; ++e2)
                s += (comb[(size_t)t * NEXP + e2] > 0.0f) ? 1 : 0;
            sl[(size_t)e * T_TOK + pos] = s;
        }
        count += total;
    }
    if (lane == 0) counts[e] = count;
}

__global__ void prefix_kernel(const int* __restrict__ counts, int* __restrict__ prefix) {
    if (threadIdx.x == 0 && blockIdx.x == 0) {
        int s = 0;
        for (int e = 0; e < NEXP; ++e) { prefix[e] = s; s += counts[e]; }
        prefix[NEXP] = s;
    }
}

// ---------------------------------------------------------------------------
// fp32 -> bf16 staging kernels
// ---------------------------------------------------------------------------
__global__ __launch_bounds__(256) void cvt_bf16(
    const float* __restrict__ src, bf16_t* __restrict__ dst, long n) {
    long i  = (long)blockIdx.x * blockDim.x + threadIdx.x;
    long st = (long)gridDim.x * blockDim.x;
    for (; i < n; i += st) dst[i] = (bf16_t)src[i];
}

// fp32 [R][C] -> bf16 [C][R] per batch (grid.z). R,C multiples of 32.
__global__ __launch_bounds__(256) void tcvt_kernel(
    const float* __restrict__ src, bf16_t* __restrict__ dst, int R, int C) {
    __shared__ float tile[32][33];
    size_t mb = (size_t)blockIdx.z * R * C;
    int c0 = blockIdx.x * 32, r0 = blockIdx.y * 32;
    int tx = threadIdx.x, ty = threadIdx.y;     // 32 x 8
#pragma unroll
    for (int i = 0; i < 4; ++i)
        tile[ty + i * 8][tx] = src[mb + (size_t)(r0 + ty + i * 8) * C + (c0 + tx)];
    __syncthreads();
#pragma unroll
    for (int i = 0; i < 4; ++i)
        dst[mb + (size_t)(c0 + ty + i * 8) * R + (r0 + tx)] =
            (bf16_t)tile[tx][ty + i * 8];
}

// ---------------------------------------------------------------------------
// Dual-B GEMM with silu(h1*h3) epilogue. B tiles via TDM, double-buffered LDS,
// software-pipelined A loads.
// ---------------------------------------------------------------------------
__global__ __launch_bounds__(256) void gemm_dual(
    const bf16_t* __restrict__ A, int K,
    const bf16_t* __restrict__ B1w, const bf16_t* __restrict__ B3w, long wstride,
    const float* __restrict__ bias1, const float* __restrict__ bias3, int bstride,
    bf16_t* __restrict__ Out, int ldo, int N,
    const int* __restrict__ idx, int idxStride,
    const int* __restrict__ counts, int Mfixed,
    const int* __restrict__ prefix) {
    int e    = blockIdx.z;
    int Mblk = counts ? counts[e] : Mfixed;
    int rb   = blockIdx.y * BM;
    if (rb >= Mblk) return;
    int cb = blockIdx.x * BN;

    const bf16_t* B1   = B1w + (size_t)e * wstride;
    const bf16_t* B3   = B3w + (size_t)e * wstride;
    const float*  b1p  = bias1 + (size_t)e * bstride;
    const float*  b3p  = bias3 + (size_t)e * bstride;
    const int*    idxp = idx ? idx + (size_t)e * idxStride : nullptr;
    int orow0 = prefix ? prefix[e] : 0;

    __shared__ __align__(16) bf16_t As [2][BM * LDT];
    __shared__ __align__(16) bf16_t Bs1[2][BN * LDT];
    __shared__ __align__(16) bf16_t Bs3[2][BN * LDT];

    int tid  = threadIdx.x;
    int lane = tid & 31, wid = tid >> 5;
    int wm = wid >> 2, wn = wid & 3;      // 2 x 4 waves
    int lr = lane & 15, kh = lane >> 4;

    int row = tid >> 2;                   // 0..63 (A tile rows)
    int seg = tid & 3;                    // 16B segment within a 64B K-row
    int  aloc   = rb + row;
    bool avalid = aloc < Mblk;
    size_t agrow = idxp ? (size_t)(avalid ? idxp[aloc] : 0) : (size_t)aloc;
    const bf16_t* aptr = A + agrow * (size_t)K + seg * 8;

    const bf16_t* b1tile = B1 + (size_t)cb * K;   // tile base (row cb, k 0)
    const bf16_t* b3tile = B3 + (size_t)cb * K;

    v8f acc1[2][2], acc3[2][2];
#pragma unroll
    for (int i = 0; i < 2; ++i)
#pragma unroll
        for (int j = 0; j < 2; ++j) { acc1[i][j] = zero_v8f(); acc3[i][j] = zero_v8f(); }

    // ---- preamble: kick off tile 0 ----
    if (wid == 0) {
        tdm_load_tile(b1tile, lds_off(&Bs1[0][0]), K, BN);
        tdm_load_tile(b3tile, lds_off(&Bs3[0][0]), K, BN);
    }
    {
        v8bf av0 = avalid ? *(const v8bf*)(aptr) : zero_v8bf();
        *(v8bf*)(&As[0][row * LDT + seg * 8]) = av0;
    }

    int buf = 0;
    for (int kt = 0; kt < K; kt += BK) {
        if (wid == 0) __builtin_amdgcn_s_wait_tensorcnt(0);   // current B tiles in LDS
        __syncthreads();                                      // + A stores visible, prev readers done

        int ktn = kt + BK;
        bool hasNext = ktn < K;
        v8bf avNext = zero_v8bf();
        if (hasNext) {
            if (wid == 0) {                                   // DMA next B tiles into other buffer
                tdm_load_tile(b1tile + ktn, lds_off(&Bs1[buf ^ 1][0]), K, BN);
                tdm_load_tile(b3tile + ktn, lds_off(&Bs3[buf ^ 1][0]), K, BN);
            }
            if (avalid) avNext = *(const v8bf*)(aptr + ktn);  // overlap with WMMAs below
        }

        v16bf af[2], bf1[2], bf3[2];
#pragma unroll
        for (int i = 0; i < 2; ++i)
            af[i] = frag16(&As[buf][(wm * 32 + i * 16 + lr) * LDT], kh);
#pragma unroll
        for (int j = 0; j < 2; ++j) {
            bf1[j] = frag16(&Bs1[buf][(wn * 32 + j * 16 + lr) * LDT], kh);
            bf3[j] = frag16(&Bs3[buf][(wn * 32 + j * 16 + lr) * LDT], kh);
        }
#pragma unroll
        for (int i = 0; i < 2; ++i)
#pragma unroll
            for (int j = 0; j < 2; ++j) {
                acc1[i][j] = wmma_bf16(af[i], bf1[j], acc1[i][j]);
                acc3[i][j] = wmma_bf16(af[i], bf3[j], acc3[i][j]);
            }

        if (hasNext)
            *(v8bf*)(&As[buf ^ 1][row * LDT + seg * 8]) = avNext;
        buf ^= 1;
    }

    // epilogue: h = (h1+b1)*(h3+b3); silu(h); store bf16
#pragma unroll
    for (int i = 0; i < 2; ++i) {
#pragma unroll
        for (int j = 0; j < 2; ++j) {
            int gc = cb + wn * 32 + j * 16 + lr;
#pragma unroll
            for (int p = 0; p < 8; ++p) {
                int grow = rb + wm * 32 + i * 16 + kh * 8 + p;
                if (grow < Mblk) {
                    float h1 = acc1[i][j][p] + b1p[gc];
                    float h3 = acc3[i][j][p] + b3p[gc];
                    float pd = h1 * h3;
                    float act = pd / (1.0f + __expf(-pd));
                    Out[(size_t)(orow0 + grow) * ldo + gc] = (bf16_t)act;
                }
            }
        }
    }
}

// ---------------------------------------------------------------------------
// Single-B GEMM (down-projection), same TDM + double-buffer pipeline.
// Expert mode: scatter w*(acc+b2) into contrib[slot][token]; shared: direct.
// ---------------------------------------------------------------------------
__global__ __launch_bounds__(256) void gemm_single(
    const bf16_t* __restrict__ A, int K,
    const bf16_t* __restrict__ Bw, long wstride,
    const float* __restrict__ bias, int bstride,
    float* __restrict__ OutBase, int N,
    const int* __restrict__ idx, const float* __restrict__ wl,
    const int* __restrict__ sl, int idxStride, int Ttot,
    const int* __restrict__ counts, int Mfixed,
    const int* __restrict__ prefix) {
    int e    = blockIdx.z;
    int Mblk = counts ? counts[e] : Mfixed;
    int rb   = blockIdx.y * BM;
    if (rb >= Mblk) return;
    int cb = blockIdx.x * BN;

    const bf16_t* B  = Bw + (size_t)e * wstride;
    const float*  bp = bias + (size_t)e * bstride;
    int arow0 = prefix ? prefix[e] : 0;

    __shared__ __align__(16) bf16_t As[2][BM * LDT];
    __shared__ __align__(16) bf16_t Bs[2][BN * LDT];

    int tid  = threadIdx.x;
    int lane = tid & 31, wid = tid >> 5;
    int wm = wid >> 2, wn = wid & 3;
    int lr = lane & 15, kh = lane >> 4;

    int row = tid >> 2;
    int seg = tid & 3;
    int  aloc   = rb + row;
    bool avalid = aloc < Mblk;
    size_t agrow = (size_t)arow0 + (size_t)(avalid ? aloc : 0);
    const bf16_t* aptr  = A + agrow * (size_t)K + seg * 8;
    const bf16_t* btile = B + (size_t)cb * K;

    v8f acc[2][2];
#pragma unroll
    for (int i = 0; i < 2; ++i)
#pragma unroll
        for (int j = 0; j < 2; ++j) acc[i][j] = zero_v8f();

    if (wid == 0)
        tdm_load_tile(btile, lds_off(&Bs[0][0]), K, BN);
    {
        v8bf av0 = avalid ? *(const v8bf*)(aptr) : zero_v8bf();
        *(v8bf*)(&As[0][row * LDT + seg * 8]) = av0;
    }

    int buf = 0;
    for (int kt = 0; kt < K; kt += BK) {
        if (wid == 0) __builtin_amdgcn_s_wait_tensorcnt(0);
        __syncthreads();

        int ktn = kt + BK;
        bool hasNext = ktn < K;
        v8bf avNext = zero_v8bf();
        if (hasNext) {
            if (wid == 0)
                tdm_load_tile(btile + ktn, lds_off(&Bs[buf ^ 1][0]), K, BN);
            if (avalid) avNext = *(const v8bf*)(aptr + ktn);
        }

        v16bf af[2], bfv[2];
#pragma unroll
        for (int i = 0; i < 2; ++i)
            af[i] = frag16(&As[buf][(wm * 32 + i * 16 + lr) * LDT], kh);
#pragma unroll
        for (int j = 0; j < 2; ++j)
            bfv[j] = frag16(&Bs[buf][(wn * 32 + j * 16 + lr) * LDT], kh);
#pragma unroll
        for (int i = 0; i < 2; ++i)
#pragma unroll
            for (int j = 0; j < 2; ++j)
                acc[i][j] = wmma_bf16(af[i], bfv[j], acc[i][j]);

        if (hasNext)
            *(v8bf*)(&As[buf ^ 1][row * LDT + seg * 8]) = avNext;
        buf ^= 1;
    }

#pragma unroll
    for (int i = 0; i < 2; ++i) {
#pragma unroll
        for (int j = 0; j < 2; ++j) {
            int gc = cb + wn * 32 + j * 16 + lr;
#pragma unroll
            for (int p = 0; p < 8; ++p) {
                int grow = rb + wm * 32 + i * 16 + kh * 8 + p;
                if (grow < Mblk) {
                    float y = acc[i][j][p] + bp[gc];
                    if (idx) {
                        size_t q = (size_t)e * idxStride + grow;
                        int t = idx[q];
                        float w = wl[q];
                        int s = sl[q];
                        OutBase[((size_t)s * Ttot + t) * N + gc] = w * y;
                    } else {
                        OutBase[(size_t)grow * N + gc] = y;
                    }
                }
            }
        }
    }
}

__global__ __launch_bounds__(256) void combine_kernel(
    const float* __restrict__ c0, const float* __restrict__ c1,
    const float* __restrict__ s, float* __restrict__ out, long n) {
    long i  = (long)blockIdx.x * blockDim.x + threadIdx.x;
    long st = (long)gridDim.x * blockDim.x;
    for (; i < n; i += st) out[i] = c0[i] + c1[i] + s[i];
}

// ---------------------------------------------------------------------------
// workspace layout
// ---------------------------------------------------------------------------
static constexpr size_t alup(size_t x) { return (x + 255) & ~(size_t)255; }
static constexpr long  T_ = T_TOK, D_ = DMODEL, F_ = DFF, E_ = NEXP, FS_ = FSHARED;

static constexpr size_t OFF_COMB   = 0;
static constexpr size_t OFF_COUNTS = alup(OFF_COMB   + (size_t)T_ * E_ * 4);
static constexpr size_t OFF_PREFIX = alup(OFF_COUNTS + 64);
static constexpr size_t OFF_IDX    = alup(OFF_PREFIX + 64);
static constexpr size_t OFF_WL     = alup(OFF_IDX    + (size_t)E_ * T_ * 4);
static constexpr size_t OFF_SL     = alup(OFF_WL     + (size_t)E_ * T_ * 4);
static constexpr size_t OFF_XB     = alup(OFF_SL     + (size_t)E_ * T_ * 4);
static constexpr size_t OFF_W1T    = alup(OFF_XB     + (size_t)T_ * D_ * 2);
static constexpr size_t OFF_W3T    = alup(OFF_W1T    + (size_t)E_ * F_ * D_ * 2);
static constexpr size_t OFF_W2T    = alup(OFF_W3T    + (size_t)E_ * F_ * D_ * 2);
static constexpr size_t OFF_WS1T   = alup(OFF_W2T    + (size_t)E_ * D_ * F_ * 2);
static constexpr size_t OFF_WS3T   = alup(OFF_WS1T   + (size_t)FS_ * D_ * 2);
static constexpr size_t OFF_WS2T   = alup(OFF_WS3T   + (size_t)FS_ * D_ * 2);
static constexpr size_t OFF_ABUF   = alup(OFF_WS2T   + (size_t)D_ * FS_ * 2);
static constexpr size_t OFF_HBUF   = alup(OFF_ABUF   + (size_t)2 * T_ * F_ * 2);
static constexpr size_t OFF_CONTR  = alup(OFF_HBUF   + (size_t)T_ * FS_ * 2);
static constexpr size_t OFF_SOUT   = alup(OFF_CONTR  + (size_t)2 * T_ * D_ * 4);

extern "C" void kernel_launch(void* const* d_in, const int* in_sizes, int n_in,
                              void* d_out, int out_size, void* d_ws, size_t ws_size,
                              hipStream_t stream) {
    (void)in_sizes; (void)n_in; (void)out_size; (void)ws_size;

    const float* x   = (const float*)d_in[0];
    const float* Wg  = (const float*)d_in[1];
    const float* W1  = (const float*)d_in[2];
    const float* b1  = (const float*)d_in[3];
    const float* W3  = (const float*)d_in[4];
    const float* b3  = (const float*)d_in[5];
    const float* W2  = (const float*)d_in[6];
    const float* b2  = (const float*)d_in[7];
    const float* Ws1 = (const float*)d_in[8];
    const float* bs1 = (const float*)d_in[9];
    const float* Ws3 = (const float*)d_in[10];
    const float* bs3 = (const float*)d_in[11];
    const float* Ws2 = (const float*)d_in[12];
    const float* bs2 = (const float*)d_in[13];
    float* out = (float*)d_out;

    char* ws = (char*)d_ws;
    float*  comb    = (float*)(ws + OFF_COMB);
    int*    counts  = (int*)  (ws + OFF_COUNTS);
    int*    prefix  = (int*)  (ws + OFF_PREFIX);
    int*    idx     = (int*)  (ws + OFF_IDX);
    float*  wl      = (float*)(ws + OFF_WL);
    int*    sl      = (int*)  (ws + OFF_SL);
    bf16_t* xb      = (bf16_t*)(ws + OFF_XB);
    bf16_t* W1t     = (bf16_t*)(ws + OFF_W1T);
    bf16_t* W3t     = (bf16_t*)(ws + OFF_W3T);
    bf16_t* W2t     = (bf16_t*)(ws + OFF_W2T);
    bf16_t* Ws1t    = (bf16_t*)(ws + OFF_WS1T);
    bf16_t* Ws3t    = (bf16_t*)(ws + OFF_WS3T);
    bf16_t* Ws2t    = (bf16_t*)(ws + OFF_WS2T);
    bf16_t* abuf    = (bf16_t*)(ws + OFF_ABUF);
    bf16_t* hbuf    = (bf16_t*)(ws + OFF_HBUF);
    float*  contrib = (float*)(ws + OFF_CONTR);
    float*  sout    = (float*)(ws + OFF_SOUT);

    // 1) gating + routing
    gate_kernel<<<T_TOK / 8, 256, 0, stream>>>(x, Wg, comb);
    build_lists<<<NEXP, 32, 0, stream>>>(comb, idx, wl, sl, counts);
    prefix_kernel<<<1, 32, 0, stream>>>(counts, prefix);

    // 2) fp32 -> bf16 staging (activations plain, weights transposed to [N][K])
    cvt_bf16<<<4096, 256, 0, stream>>>(x, xb, (long)T_ * D_);
    dim3 tb(32, 8);
    tcvt_kernel<<<dim3(F_ / 32, D_ / 32, NEXP), tb, 0, stream>>>(W1, W1t, DMODEL, DFF);
    tcvt_kernel<<<dim3(F_ / 32, D_ / 32, NEXP), tb, 0, stream>>>(W3, W3t, DMODEL, DFF);
    tcvt_kernel<<<dim3(D_ / 32, F_ / 32, NEXP), tb, 0, stream>>>(W2, W2t, DFF, DMODEL);
    tcvt_kernel<<<dim3(FS_ / 32, D_ / 32, 1), tb, 0, stream>>>(Ws1, Ws1t, DMODEL, FSHARED);
    tcvt_kernel<<<dim3(FS_ / 32, D_ / 32, 1), tb, 0, stream>>>(Ws3, Ws3t, DMODEL, FSHARED);
    tcvt_kernel<<<dim3(D_ / 32, FS_ / 32, 1), tb, 0, stream>>>(Ws2, Ws2t, FSHARED, DMODEL);

    // 3) expert up-projection over gathered tokens
    gemm_dual<<<dim3(DFF / BN, T_TOK / BM, NEXP), 256, 0, stream>>>(
        xb, DMODEL, W1t, W3t, (long)F_ * D_, b1, b3, DFF,
        abuf, DFF, DFF, idx, T_TOK, counts, 0, prefix);

    // 4) shared up-projection
    gemm_dual<<<dim3(FSHARED / BN, T_TOK / BM, 1), 256, 0, stream>>>(
        xb, DMODEL, Ws1t, Ws3t, 0, bs1, bs3, 0,
        hbuf, FSHARED, FSHARED, nullptr, 0, nullptr, T_TOK, nullptr);

    // 5) expert down-projection, gate-weighted deterministic scatter
    gemm_single<<<dim3(DMODEL / BN, T_TOK / BM, NEXP), 256, 0, stream>>>(
        abuf, DFF, W2t, (long)D_ * F_, b2, DMODEL,
        contrib, DMODEL, idx, wl, sl, T_TOK, T_TOK, counts, 0, prefix);

    // 6) shared down-projection
    gemm_single<<<dim3(DMODEL / BN, T_TOK / BM, 1), 256, 0, stream>>>(
        hbuf, FSHARED, Ws2t, 0, bs2, 0,
        sout, DMODEL, nullptr, nullptr, nullptr, 0, T_TOK, nullptr, T_TOK, nullptr);

    // 7) final sum (fixed order => deterministic)
    combine_kernel<<<4096, 256, 0, stream>>>(
        contrib, contrib + (size_t)T_ * D_, sout, out, (long)T_ * D_);
}
